// QuantumSSMCore_83863531421711
// MI455X (gfx1250) — matrix operand order repaired and checked by reading, not compiled, not run on hardware
//
#include <hip/hip_runtime.h>
#include <math.h>

typedef unsigned short u16;
typedef __attribute__((ext_vector_type(16))) __bf16 v16bf;
typedef __attribute__((ext_vector_type(16))) u16    v16u;
typedef __attribute__((ext_vector_type(8)))  float  v8f;
typedef __attribute__((ext_vector_type(4)))  int    i4;

#define BATCH  8
#define SEQ    2048
#define DMODEL 768
#define MTOK   (BATCH*SEQ)   /* 16384 */
#define PI_F   3.14159265358979323846f

/* ---- gfx1250 async-to-LDS path (guarded; falls back to sync staging) ---- */
#if defined(__has_builtin)
#if __has_builtin(__builtin_amdgcn_global_load_async_to_lds_b128) && \
    __has_builtin(__builtin_amdgcn_s_wait_asynccnt)
#define HAVE_ASYNC 1
#endif
#endif
#ifndef HAVE_ASYNC
#define HAVE_ASYNC 0
#endif

/* ---- workspace layout (bytes, 256-aligned); all B matrices pre-transposed ---- */
#define OFF_XBF   ((size_t)0)         /* 16384x768 bf16  = 25165824 */
#define OFF_WAT   ((size_t)25165824)  /* 64x768  bf16 (Wa^T)   =   98304 */
#define OFF_WDTT  ((size_t)25264128)  /* 768x64  bf16 (Wdt^T)  =   98304 */
#define OFF_W3T   ((size_t)25362432)  /* 768x768 bf16 (W3^T)   = 1179648 */
#define OFF_W2T   ((size_t)26542080)  /* 768x32  bf16 (W2^T)   =   49152 */
#define OFF_BIAS2 ((size_t)26591232)  /* 768 f32               =    3072 */
#define OFF_DRBF  ((size_t)26594304)  /* 16384x64 bf16         = 2097152 */
#define OFF_XQ    ((size_t)28691456)  /* 16384x6 f32           =  393216 */
#define OFF_PART  ((size_t)29084672)  /* 16384x12 f32          =  786432 */
#define OFF_DTS   ((size_t)29871104)  /* 16384 f32             =   65536 */
#define OFF_QBF   ((size_t)29936640)  /* 16384x32 bf16         = 1048576 */

__device__ __forceinline__ u16 f2bf(float f) {
    unsigned int u = __float_as_uint(f);
    u += 0x7FFFu + ((u >> 16) & 1u);       /* RNE */
    return (u16)(u >> 16);
}

/* ================= prep kernels ================= */

__global__ __launch_bounds__(256) void k_prep1(
    const float* __restrict__ x_proj_w, const float* __restrict__ dt_proj_w,
    const float* __restrict__ qin_w, const float* __restrict__ Dvec,
    const float* __restrict__ out_w,
    u16* __restrict__ Wat, u16* __restrict__ Wdtt, u16* __restrict__ W3t)
{
    int i = blockIdx.x * 256 + threadIdx.x;
    if (i < DMODEL * DMODEL) {             /* W3t[n][k] = D[k]*out_w[k][n] */
        int n = i / DMODEL, k = i % DMODEL;
        W3t[i] = f2bf(Dvec[k] * out_w[(size_t)k * DMODEL + n]);
    }
    if (i < 64 * DMODEL) {                 /* Wat[n][k]: n<64, k<768 */
        int n = i / DMODEL, k = i % DMODEL;
        float v = 0.f;
        if (n < 48)      v = x_proj_w[k * 80 + n];
        else if (n < 54) v = qin_w[k * 6 + (n - 48)];
        Wat[i] = f2bf(v);
    }
    if (i < DMODEL * 64) {                 /* Wdtt[n][kk]: n<768, kk<64 (48 real) */
        int n = i >> 6, kk = i & 63;
        Wdtt[i] = (kk < 48) ? f2bf(dt_proj_w[kk * DMODEL + n]) : (u16)0;
    }
}

__global__ __launch_bounds__(256) void k_prep2(
    const float* __restrict__ qout_w, const float* __restrict__ qout_b,
    const float* __restrict__ out_w, const float* __restrict__ out_b,
    u16* __restrict__ W2t, float* __restrict__ bias2)
{
    int n = blockIdx.x * 256 + threadIdx.x;
    if (n >= DMODEL) return;
    float acc[18];
#pragma unroll
    for (int j = 0; j < 18; ++j) acc[j] = 0.f;
    float ab = 0.f;
    for (int k = 0; k < DMODEL; ++k) {
        float ow = out_w[(size_t)k * DMODEL + n];
#pragma unroll
        for (int j = 0; j < 18; ++j) acc[j] += qout_w[j * DMODEL + k] * ow;
        ab += qout_b[k] * ow;
    }
#pragma unroll
    for (int j = 0; j < 18; ++j) W2t[n * 32 + j] = f2bf(acc[j]);
#pragma unroll
    for (int j = 18; j < 32; ++j) W2t[n * 32 + j] = 0;
    bias2[n] = ab + out_b[n];
}

__global__ __launch_bounds__(256) void k_convert(
    const float* __restrict__ x, u16* __restrict__ Xbf,
    u16* __restrict__ Qbf, u16* __restrict__ DRbf)
{
    int i = blockIdx.x * 256 + threadIdx.x;             /* < 16384*768 */
    Xbf[i] = f2bf(x[i]);
    if (i < MTOK * 32) Qbf[i]  = 0;                     /* pad cols of Q */
    if (i < MTOK * 64) DRbf[i] = 0;                     /* pad cols of delta_raw */
}

/* ================= WMMA GEMM core =================
 * 256 thr (8 waves); tile 128(M) x 64(N); K-step 32, bf16; B pre-transposed. */

__device__ __forceinline__ v16bf lds_afrag(const u16 (*lA)[40], int mbase, int lane)
{
    int row = mbase + (lane & 15);
    int kb  = (lane >> 4) * 8;           /* lanes 0-15: K 0..7,16..23 ; 16-31: 8..15,24..31 */
    v16u a;
    ((uint4*)&a)[0] = *(const uint4*)&lA[row][kb];
    ((uint4*)&a)[1] = *(const uint4*)&lA[row][kb + 16];
    return __builtin_bit_cast(v16bf, a);
}

__device__ __forceinline__ v16bf lds_bfrag(const u16 (*lBt)[40], int nbase, int lane)
{
    int col = nbase + (lane & 15);
    int kb  = (lane >> 4) * 8;           /* lanes 0-15: K 0..7,16..23 ; 16-31: 8..15,24..31 */
    v16u a;
    ((uint4*)&a)[0] = *(const uint4*)&lBt[col][kb];
    ((uint4*)&a)[1] = *(const uint4*)&lBt[col][kb + 16];
    return __builtin_bit_cast(v16bf, a);
}

#if HAVE_ASYNC
__device__ __forceinline__ void async_b128(const u16* g, u16* l) {
    __builtin_amdgcn_global_load_async_to_lds_b128(
        (__attribute__((address_space(1))) i4*)(void*)g,
        (__attribute__((address_space(3))) i4*)(void*)l,
        0, 0);
}
#endif

/* stage one 128x32 A tile + one 64x32 B^T tile */
__device__ __forceinline__ void issue_tile(
    const u16* __restrict__ A, int lda, const u16* __restrict__ Bt, int ldb,
    int m0, int n0, int k, u16 (*lA)[40], u16 (*lBt)[40], int tid)
{
    int arow = tid >> 1, ahalf = (tid & 1) * 16;
    const u16* ga = A + (size_t)(m0 + arow) * lda + k + ahalf;
    int brow = tid >> 2, bchunk = (tid & 3) * 8;
    const u16* gb = Bt + (size_t)(n0 + brow) * ldb + k + bchunk;
#if HAVE_ASYNC
    async_b128(ga,     &lA[arow][ahalf]);
    async_b128(ga + 8, &lA[arow][ahalf + 8]);
    async_b128(gb,     &lBt[brow][bchunk]);
#else
    *(uint4*)&lA[arow][ahalf]     = *(const uint4*)ga;
    *(uint4*)&lA[arow][ahalf + 8] = *(const uint4*)(ga + 8);
    *(uint4*)&lBt[brow][bchunk]   = *(const uint4*)gb;
    __builtin_prefetch(ga + 32, 0, 1);
#endif
}

__device__ __forceinline__ void compute_tile(
    const u16 (*lA)[40], const u16 (*lBt)[40], int lane, int wave, v8f acc[4])
{
    v16bf af = lds_afrag(lA, wave * 16, lane);
#pragma unroll
    for (int nt = 0; nt < 4; ++nt) {
        v16bf bfr = lds_bfrag(lBt, nt * 16, lane);
        acc[nt] = __builtin_amdgcn_wmma_f32_16x16x32_bf16(
            false, af, false, bfr, (short)0, acc[nt], false, false);
    }
}

/* two-phase K loop (phase1 may be empty: K1==0) */
__device__ __forceinline__ void gemm_body(
    const u16* A0, int lda0, const u16* B0, int ldb0, int K0,
    const u16* A1, int lda1, const u16* B1, int ldb1, int K1,
    int m0, int n0, u16 (*lA)[128][40], u16 (*lBt)[64][40],
    int tid, int lane, int wave, v8f acc[4])
{
    const int nt0 = K0 / 32, NT = nt0 + K1 / 32;
    auto src = [&](int kt, const u16*& A, int& lda, const u16*& B, int& ldb, int& k) {
        if (kt < nt0) { A = A0; lda = lda0; B = B0; ldb = ldb0; k = kt * 32; }
        else          { A = A1; lda = lda1; B = B1; ldb = ldb1; k = (kt - nt0) * 32; }
    };
#if HAVE_ASYNC
    {   const u16 *A, *B; int lda, ldb, k;
        src(0, A, lda, B, ldb, k);
        issue_tile(A, lda, B, ldb, m0, n0, k, lA[0], lBt[0], tid);
    }
    for (int kt = 0; kt < NT; ++kt) {
        __builtin_amdgcn_s_wait_asynccnt(0);   /* own tile-kt loads landed in LDS */
        __syncthreads();                       /* all waves' tile-kt data visible  */
        if (kt + 1 < NT) {                     /* overlap next-tile DMA w/ compute */
            const u16 *A, *B; int lda, ldb, k;
            src(kt + 1, A, lda, B, ldb, k);
            issue_tile(A, lda, B, ldb, m0, n0, k, lA[(kt + 1) & 1], lBt[(kt + 1) & 1], tid);
        }
        compute_tile((const u16(*)[40])lA[kt & 1], (const u16(*)[40])lBt[kt & 1],
                     lane, wave, acc);
    }
#else
    for (int kt = 0; kt < NT; ++kt) {
        const u16 *A, *B; int lda, ldb, k;
        src(kt, A, lda, B, ldb, k);
        __syncthreads();
        issue_tile(A, lda, B, ldb, m0, n0, k, lA[0], lBt[0], tid);
        __syncthreads();
        compute_tile((const u16(*)[40])lA[0], (const u16(*)[40])lBt[0], lane, wave, acc);
    }
#endif
}

/* GEMM1: Xbf(16384x768) @ Wa(768x64) -> delta_raw bf16 (cols<48) + xq (cols 48..53) */
__global__ __launch_bounds__(256) void k_gemm1(
    const u16* __restrict__ Xbf, const u16* __restrict__ Wat,
    const float* __restrict__ qin_b,
    u16* __restrict__ DRbf, float* __restrict__ xq)
{
    __shared__ __align__(16) u16 lA[2][128][40];
    __shared__ __align__(16) u16 lBt[2][64][40];
    int tid = threadIdx.x, lane = tid & 31, wave = tid >> 5;
    int m0 = blockIdx.x * 128;
    v8f acc[4] = {};
    gemm_body(Xbf, DMODEL, Wat, DMODEL, DMODEL,
              Xbf, DMODEL, Wat, DMODEL, 0,
              m0, 0, lA, lBt, tid, lane, wave, acc);
#pragma unroll
    for (int nt = 0; nt < 4; ++nt)
#pragma unroll
        for (int v = 0; v < 8; ++v) {
            int m = m0 + wave * 16 + v + 8 * (lane >> 4);
            int n = nt * 16 + (lane & 15);
            float val = acc[nt][v];
            if (n < 48)      DRbf[(size_t)m * 64 + n] = f2bf(val);
            else if (n < 54) xq[(size_t)m * 6 + (n - 48)] = tanhf(val + qin_b[n - 48]) * PI_F;
        }
}

/* GEMM2: DRbf(16384x64) @ Wdt(64x768) + b -> softplus -> per-row partial sums */
__global__ __launch_bounds__(256) void k_gemm2(
    const u16* __restrict__ DRbf, const u16* __restrict__ Wdtt,
    const float* __restrict__ dt_proj_b, float* __restrict__ part)
{
    __shared__ __align__(16) u16 lA[2][128][40];
    __shared__ __align__(16) u16 lBt[2][64][40];
    int tid = threadIdx.x, lane = tid & 31, wave = tid >> 5;
    int m0 = blockIdx.x * 128, n0 = blockIdx.y * 64;
    v8f acc[4] = {};
    gemm_body(DRbf, 64, Wdtt, 64, 64,
              DRbf, 64, Wdtt, 64, 0,
              m0, n0, lA, lBt, tid, lane, wave, acc);
    float rs[8];
#pragma unroll
    for (int v = 0; v < 8; ++v) rs[v] = 0.f;
#pragma unroll
    for (int nt = 0; nt < 4; ++nt)
#pragma unroll
        for (int v = 0; v < 8; ++v) {
            int n = n0 + nt * 16 + (lane & 15);
            float val = acc[nt][v] + dt_proj_b[n];
            rs[v] += (val > 20.f) ? val : log1pf(__expf(val));   /* softplus */
        }
#pragma unroll
    for (int v = 0; v < 8; ++v) {
        float s = rs[v];
#pragma unroll
        for (int mm = 1; mm < 16; mm <<= 1) s += __shfl_xor(s, mm, 32);
        if ((lane & 15) == 0) {
            int m = m0 + wave * 16 + v + 8 * (lane >> 4);
            part[(size_t)m * 12 + blockIdx.y] = s;   /* exactly one writer */
        }
    }
}

__global__ __launch_bounds__(256) void k_dtsig(
    const float* __restrict__ part, float* __restrict__ dts)
{
    int i = blockIdx.x * 256 + threadIdx.x;
    if (i >= MTOK) return;
    float s = 0.f;
#pragma unroll
    for (int j = 0; j < 12; ++j) s += part[(size_t)i * 12 + j];
    dts[i] = 1.f / (1.f + __expf(-s * (1.f / 768.f)));
}

/* ================= quantum scan: one wave32 per batch ================= */

__device__ __forceinline__ void g_ry(int t, float c, float s, int L,
    float& a0r, float& a0i, float& a1r, float& a1i)
{
    if (t == 0) {
        float n0r = c*a0r - s*a1r, n0i = c*a0i - s*a1i;
        float n1r = s*a0r + c*a1r, n1i = s*a0i + c*a1i;
        a0r=n0r; a0i=n0i; a1r=n1r; a1i=n1i;
    } else {
        int m = 1 << (t - 1);
        float p0r=__shfl_xor(a0r,m,32), p0i=__shfl_xor(a0i,m,32);
        float p1r=__shfl_xor(a1r,m,32), p1i=__shfl_xor(a1i,m,32);
        float sg = ((L >> (t-1)) & 1) ? s : -s;
        a0r = c*a0r + sg*p0r;  a0i = c*a0i + sg*p0i;
        a1r = c*a1r + sg*p1r;  a1i = c*a1i + sg*p1i;
    }
}

__device__ __forceinline__ void g_rx(int t, float c, float s, int L,
    float& a0r, float& a0i, float& a1r, float& a1i)
{
    if (t == 0) {
        float n0r = c*a0r + s*a1i, n0i = c*a0i - s*a1r;
        float n1r = c*a1r + s*a0i, n1i = c*a1i - s*a0r;
        a0r=n0r; a0i=n0i; a1r=n1r; a1i=n1i;
    } else {
        int m = 1 << (t - 1);
        float p0r=__shfl_xor(a0r,m,32), p0i=__shfl_xor(a0i,m,32);
        float p1r=__shfl_xor(a1r,m,32), p1i=__shfl_xor(a1i,m,32);
        a0r = c*a0r + s*p0i;  a0i = c*a0i - s*p0r;   /* RX symmetric */
        a1r = c*a1r + s*p1i;  a1i = c*a1i - s*p1r;
    }
}

__device__ __forceinline__ void g_rz(int t, float c, float s, int L,
    float& a0r, float& a0i, float& a1r, float& a1i)
{
    if (t == 0) {
        float n0r = c*a0r + s*a0i, n0i = c*a0i - s*a0r;   /* *(c - i s) */
        float n1r = c*a1r - s*a1i, n1i = c*a1i + s*a1r;   /* *(c + i s) */
        a0r=n0r; a0i=n0i; a1r=n1r; a1i=n1i;
    } else {
        float sg = ((L >> (t-1)) & 1) ? -s : s;
        float n0r = c*a0r + sg*a0i, n0i = c*a0i - sg*a0r;
        float n1r = c*a1r + sg*a1i, n1i = c*a1i - sg*a1r;
        a0r=n0r; a0i=n0i; a1r=n1r; a1i=n1i;
    }
}

__device__ __forceinline__ void g_crx(int cq, int t, float c, float s, int L,
    float& a0r, float& a0i, float& a1r, float& a1i)
{
    if (cq == 0) {                         /* only r==1 amplitudes (A1); t>=1 here */
        int m = 1 << (t - 1);
        float p1r=__shfl_xor(a1r,m,32), p1i=__shfl_xor(a1i,m,32);
        a1r = c*a1r + s*p1i;  a1i = c*a1i - s*p1r;
    } else {
        bool act = (L >> (cq - 1)) & 1;
        if (t == 0) {
            if (act) {
                float n0r = c*a0r + s*a1i, n0i = c*a0i - s*a1r;
                float n1r = c*a1r + s*a0i, n1i = c*a1i - s*a0r;
                a0r=n0r; a0i=n0i; a1r=n1r; a1i=n1i;
            }
        } else {
            int m = 1 << (t - 1);
            float p0r=__shfl_xor(a0r,m,32), p0i=__shfl_xor(a0i,m,32);
            float p1r=__shfl_xor(a1r,m,32), p1i=__shfl_xor(a1i,m,32);
            if (act) {
                a0r = c*a0r + s*p0i;  a0i = c*a0i - s*p0r;
                a1r = c*a1r + s*p1i;  a1i = c*a1i - s*p1r;
            }
        }
    }
}

__global__ __launch_bounds__(32) void k_scan(
    const float* __restrict__ xq, const float* __restrict__ dts,
    const float* __restrict__ cparams,
    u16* __restrict__ Qbf, float* __restrict__ hfin)
{
    const int b = blockIdx.x, L = threadIdx.x;
    const int kind = L / 6, wm = L % 6;      /* lane -> (X|Y|Z, qubit) output slot */

    float p[60];
#pragma unroll
    for (int i = 0; i < 60; ++i) p[i] = cparams[i];

    /* CRX angles do not depend on dt: hoist sin/cos out of the time loop */
    float cxc[24], cxs[24];
#pragma unroll
    for (int l = 0; l < 2; ++l) {
#pragma unroll
        for (int i = 0; i < 6; ++i)
            __sincosf(p[l*30 + 18 + i] * 0.5f, &cxs[l*12 + i], &cxc[l*12 + i]);
#pragma unroll
        for (int i2 = 0; i2 < 6; ++i2)
            __sincosf(p[l*30 + 24 + i2] * 0.5f, &cxs[l*12 + 6 + i2], &cxc[l*12 + 6 + i2]);
    }

    float h[6];
#pragma unroll
    for (int i = 0; i < 6; ++i) h[i] = 0.f;

    for (int t = 0; t < SEQ; ++t) {
        const size_t row = (size_t)b * SEQ + t;
        const float dt = dts[row];
        float ia[6];
#pragma unroll
        for (int i = 0; i < 6; ++i) ia[i] = xq[row * 6 + i];

        /* |0...0> */
        float a0r = (L == 0) ? 1.f : 0.f, a0i = 0.f, a1r = 0.f, a1i = 0.f;

#pragma unroll
        for (int i = 0; i < 6; ++i) {
            float s, c; __sincosf(h[i] * 0.5f, &s, &c);
            g_ry(i, c, s, L, a0r, a0i, a1r, a1i);
        }
#pragma unroll
        for (int l = 0; l < 2; ++l) {
            const int base = l * 30;
#pragma unroll
            for (int i = 0; i < 6; ++i) {
                float s, c;
                __sincosf(p[base + 3*i + 0] * dt * 0.5f, &s, &c); g_rx(i, c, s, L, a0r,a0i,a1r,a1i);
                __sincosf(p[base + 3*i + 1] * dt * 0.5f, &s, &c); g_ry(i, c, s, L, a0r,a0i,a1r,a1i);
                __sincosf(p[base + 3*i + 2] * dt * 0.5f, &s, &c); g_rz(i, c, s, L, a0r,a0i,a1r,a1i);
            }
#pragma unroll
            for (int i = 0; i < 6; ++i)
                g_crx(i, (i + 1) % 6, cxc[l*12 + i], cxs[l*12 + i], L, a0r,a0i,a1r,a1i);
#pragma unroll
            for (int i2 = 0; i2 < 6; ++i2) {
                int i = 5 - i2;
                g_crx(i, (i + 5) % 6, cxc[l*12 + 6 + i2], cxs[l*12 + 6 + i2], L, a0r,a0i,a1r,a1i);
            }
        }
#pragma unroll
        for (int i = 0; i < 6; ++i) {
            float s, c; __sincosf(ia[i] * dt * 0.5f, &s, &c);
            g_ry(i, c, s, L, a0r, a0i, a1r, a1i);
        }

        /* expectation values X,Y,Z per qubit; all lanes end with reduced values */
        float myv = 0.f;
#pragma unroll
        for (int w = 0; w < 6; ++w) {
            float cr, ci, zz;
            if (w == 0) {
                cr = a0r*a1r + a0i*a1i;
                ci = a0r*a1i - a0i*a1r;
                zz = (a0r*a0r + a0i*a0i) - (a1r*a1r + a1i*a1i);
            } else {
                int m = 1 << (w - 1);
                bool hi = (L >> (w - 1)) & 1;
                float p0r=__shfl_xor(a0r,m,32), p0i=__shfl_xor(a0i,m,32);
                float p1r=__shfl_xor(a1r,m,32), p1i=__shfl_xor(a1i,m,32);
                float nrm = a0r*a0r + a0i*a0i + a1r*a1r + a1i*a1i;
                if (!hi) {
                    cr = a0r*p0r + a0i*p0i + a1r*p1r + a1i*p1i;
                    ci = a0r*p0i - a0i*p0r + a1r*p1i - a1i*p1r;
                    zz = nrm;
                } else { cr = 0.f; ci = 0.f; zz = -nrm; }
            }
#pragma unroll
            for (int mm = 1; mm < 32; mm <<= 1) {
                cr += __shfl_xor(cr, mm, 32);
                ci += __shfl_xor(ci, mm, 32);
                zz += __shfl_xor(zz, mm, 32);
            }
            if (wm == w) myv = (kind == 0) ? 2.f*cr : ((kind == 1) ? 2.f*ci : zz);
            h[w] = tanhf(zz) * PI_F;          /* next-step hidden angle */
        }
        if (L < 18) Qbf[row * 32 + L] = f2bf(myv);
    }
    if (L == 0) {
#pragma unroll
        for (int i = 0; i < 6; ++i) hfin[b * 6 + i] = h[i];
    }
}

/* GEMM3: y = Xbf@W3 + Qbf@W2 + bias2 */
__global__ __launch_bounds__(256) void k_gemm3(
    const u16* __restrict__ Xbf, const u16* __restrict__ W3t,
    const u16* __restrict__ Qbf, const u16* __restrict__ W2t,
    const float* __restrict__ bias2, float* __restrict__ y)
{
    __shared__ __align__(16) u16 lA[2][128][40];
    __shared__ __align__(16) u16 lBt[2][64][40];
    int tid = threadIdx.x, lane = tid & 31, wave = tid >> 5;
    int m0 = blockIdx.x * 128, n0 = blockIdx.y * 64;
    v8f acc[4] = {};
    gemm_body(Xbf, DMODEL, W3t, DMODEL, DMODEL,
              Qbf, 32,     W2t, 32,     32,
              m0, n0, lA, lBt, tid, lane, wave, acc);
#pragma unroll
    for (int nt = 0; nt < 4; ++nt)
#pragma unroll
        for (int v = 0; v < 8; ++v) {
            int m = m0 + wave * 16 + v + 8 * (lane >> 4);
            int n = n0 + nt * 16 + (lane & 15);
            y[(size_t)m * DMODEL + n] = acc[nt][v] + bias2[n];
        }
}

/* ================= host launcher ================= */

extern "C" void kernel_launch(void* const* d_in, const int* in_sizes, int n_in,
                              void* d_out, int out_size, void* d_ws, size_t ws_size,
                              hipStream_t stream)
{
    (void)in_sizes; (void)n_in; (void)out_size; (void)ws_size;
    const float* x         = (const float*)d_in[0];
    const float* x_proj_w  = (const float*)d_in[1];
    const float* dt_proj_w = (const float*)d_in[2];
    const float* dt_proj_b = (const float*)d_in[3];
    const float* qin_w     = (const float*)d_in[4];
    const float* qin_b     = (const float*)d_in[5];
    const float* cparams   = (const float*)d_in[6];
    const float* qout_w    = (const float*)d_in[7];
    const float* qout_b    = (const float*)d_in[8];
    const float* Dvec      = (const float*)d_in[9];
    const float* out_w     = (const float*)d_in[10];
    const float* out_b     = (const float*)d_in[11];

    char* ws = (char*)d_ws;
    u16*   Xbf   = (u16*)  (ws + OFF_XBF);
    u16*   Wat   = (u16*)  (ws + OFF_WAT);
    u16*   Wdtt  = (u16*)  (ws + OFF_WDTT);
    u16*   W3t   = (u16*)  (ws + OFF_W3T);
    u16*   W2t   = (u16*)  (ws + OFF_W2T);
    float* bias2 = (float*)(ws + OFF_BIAS2);
    u16*   DRbf  = (u16*)  (ws + OFF_DRBF);
    float* xq    = (float*)(ws + OFF_XQ);
    float* part  = (float*)(ws + OFF_PART);
    float* dts   = (float*)(ws + OFF_DTS);
    u16*   Qbf   = (u16*)  (ws + OFF_QBF);

    float* y    = (float*)d_out;
    float* hfin = y + (size_t)MTOK * DMODEL;

    k_prep1<<<(DMODEL * DMODEL + 255) / 256, 256, 0, stream>>>(
        x_proj_w, dt_proj_w, qin_w, Dvec, out_w, Wat, Wdtt, W3t);
    k_prep2<<<3, 256, 0, stream>>>(qout_w, qout_b, out_w, out_b, W2t, bias2);
    k_convert<<<(MTOK * DMODEL) / 256, 256, 0, stream>>>(x, Xbf, Qbf, DRbf);

    k_gemm1<<<dim3(MTOK / 128), 256, 0, stream>>>(Xbf, Wat, qin_b, DRbf, xq);
    k_gemm2<<<dim3(MTOK / 128, DMODEL / 64), 256, 0, stream>>>(DRbf, Wdtt, dt_proj_b, part);
    k_dtsig<<<MTOK / 256, 256, 0, stream>>>(part, dts);

    k_scan<<<BATCH, 32, 0, stream>>>(xq, dts, cparams, Qbf, hfin);

    k_gemm3<<<dim3(MTOK / 128, DMODEL / 64), 256, 0, stream>>>(Xbf, W3t, Qbf, W2t, bias2, y);
}
